// FactorActorCritic_9947144257737
// MI455X (gfx1250) — matrix operand order repaired and checked by reading, not compiled, not run on hardware
//
#include <hip/hip_runtime.h>

// ---------------- problem constants (from reference) ----------------
constexpr int T_ = 4, B_ = 4, N_ = 16384, KALL_ = 4, F_ = 16384, ORD_ = 3;
constexpr int DIN_ = 4, H_ = 64, NH_ = 4, HD_ = 16, DE_ = 4, DOUT_ = 64;
constexpr int N0_ = N_ / KALL_;   // 4096 nodes per partition
constexpr int G3_ = 3 * H_;       // 192 (GRU gate width)

typedef __bf16 bf16;
typedef __attribute__((ext_vector_type(16))) __bf16 bhalf16;
typedef __attribute__((ext_vector_type(8)))  float  float8;
typedef __attribute__((address_space(3))) bf16 lds_bf16;

#define DEV static __device__ __forceinline__

// ---------------- swizzled fragment storage ----------------
// All WMMA operands live in LDS in "fragment order": for each 16x32 (A) or
// 32x16 (B) tile, lane L's 16 bf16 values are contiguous at (lane*16),
// 32-byte aligned, so a fragment load is one 32B LDS read (2x ds_load_b128).

// A panel: 16 rows x K cols (K multiple of 32). Element (m, k):
DEV int a_swz(int m, int k) {
  int kt = k >> 5, kw = k & 31;
  int ln = m + (((kw >> 3) & 1) << 4);       // bit3 of kw selects lane half
  int s  = (kw & 7) + ((kw >> 4) << 3);      // bit4 selects slot half
  return kt * 512 + ln * 16 + s;
}
DEV bhalf16 ldsA_swz(const bf16* panel, int kt, int lane) {
  return *(const bhalf16*)(panel + kt * 512 + lane * 16);
}

// B matrix: K x ncols. bit3 of k selects lane half, bit4 selects slot half (same as A).
DEV int b_swz(int k, int n, int ncols) {
  int kt = k >> 5, kw = k & 31;
  int ct = n >> 4;
  int ln = (n & 15) + (((kw >> 3) & 1) << 4);
  int s  = (kw & 7) + ((kw >> 4) << 3);
  return (kt * (ncols >> 4) + ct) * 512 + ln * 16 + s;
}
DEV bhalf16 ldsB_swz(const bf16* mat, int kt, int ct, int ncols, int lane) {
  return *(const bhalf16*)(mat + (kt * (ncols >> 4) + ct) * 512 + lane * 16);
}

DEV float8 wmma_bf16(bhalf16 a, bhalf16 b, float8 c) {
  return __builtin_amdgcn_wmma_f32_16x16x32_bf16(
      /*neg_a=*/false, a, /*neg_b=*/false, b,
      /*c_mod=*/(short)0, c, /*reuse_a=*/false, /*reuse_b=*/false);
}

DEV float8 f8zero() { float8 z = {0.f,0.f,0.f,0.f,0.f,0.f,0.f,0.f}; return z; }
DEV float sigmoidf_(float v) { return 1.f / (1.f + __expf(-v)); }

// ---------------- kernel 0: zero scatter accumulators ----------------
__global__ void zero_kernel(float* __restrict__ p, size_t n) {
  size_t i = (size_t)blockIdx.x * blockDim.x + threadIdx.x;
  size_t stride = (size_t)gridDim.x * blockDim.x;
  for (; i < n; i += stride) p[i] = 0.f;
}

// ---------------- kernel 1: GRU scan (16 rows / block) ----------------
__global__ __launch_bounds__(256)
void gru_kernel(const float* __restrict__ h0, const float* __restrict__ x,
                const unsigned char* __restrict__ dones,
                const float* __restrict__ Wi, const float* __restrict__ bi,
                const float* __restrict__ Wh, const float* __restrict__ bhn,
                bf16* __restrict__ hs, float* __restrict__ hfin) {
  __shared__ alignas(32) bf16  sWh[H_ * G3_];  // swizzled B, 24 KB
  __shared__ float sWi[DIN_ * G3_];
  __shared__ float sBi[G3_];
  __shared__ float sBhn[H_];
  __shared__ float sC[16 * H_];                // f32 carry (linear)
  __shared__ alignas(32) bf16  sCb[16 * H_];   // swizzled A panel
  __shared__ float sGh[16 * G3_];              // gh result (linear)

  const int tid = threadIdx.x, lane = tid & 31, wave = tid >> 5;
  const int row0 = blockIdx.x * 16;
  const int bb = row0 / N_;                    // batch uniform per block

  for (int i = tid; i < H_ * G3_; i += 256) {
    int k = i / G3_, n = i % G3_;
    sWh[b_swz(k, n, G3_)] = (bf16)Wh[i];
  }
  for (int i = tid; i < DIN_ * G3_; i += 256) sWi[i] = Wi[i];
  for (int i = tid; i < G3_; i += 256) sBi[i] = bi[i];
  for (int i = tid; i < H_; i += 256) sBhn[i] = bhn[i];

#pragma unroll
  for (int i = 0; i < 4; ++i) {                // 16*64 elems / 256 thr
    int e = tid + 256 * i;
    int r = e >> 6, c = e & 63;
    sC[e] = h0[(size_t)(row0 + r) * H_ + c];
  }

  for (int t = 0; t < T_; ++t) {
    const bool done = dones[t * B_ + bb] != 0;
#pragma unroll
    for (int i = 0; i < 4; ++i) {              // done-mask + swizzled bf16 copy
      int e = tid + 256 * i;
      int r = e >> 6, c = e & 63;
      float cv = done ? 0.f : sC[e];
      sC[e] = cv;
      sCb[a_swz(r, c)] = (bf16)cv;
    }
    __syncthreads();
    // gh = c @ Wh : (16x64)@(64x192), 12 col-tiles, 2 K-steps each
    for (int j = wave; j < 12; j += 8) {
      float8 acc = f8zero();
#pragma unroll
      for (int ks = 0; ks < 64; ks += 32) {
        bhalf16 a = ldsA_swz(sCb, ks >> 5, lane);
        bhalf16 b = ldsB_swz(sWh, ks >> 5, j, G3_, lane);
        acc = wmma_bf16(a, b, acc);
      }
      int n = lane & 15, mb = (lane < 16) ? 0 : 8;
#pragma unroll
      for (int i = 0; i < 8; ++i)
        sGh[(mb + i) * G3_ + j * 16 + n] = acc[i];
    }
    __syncthreads();
    // gates
#pragma unroll
    for (int i = 0; i < 4; ++i) {
      int e = tid + 256 * i;
      int r = e >> 6, c = e & 63;
      int gr = row0 + r;
      int nn = gr % N_;
      const float* xr = x + (((size_t)t * B_ + bb) * N_ + nn) * DIN_;
      float gir = sBi[c], giz = sBi[64 + c], gin = sBi[128 + c];
#pragma unroll
      for (int d = 0; d < DIN_; ++d) {
        float xv = xr[d];
        gir += xv * sWi[d * G3_ + c];
        giz += xv * sWi[d * G3_ + 64 + c];
        gin += xv * sWi[d * G3_ + 128 + c];
      }
      float cv = sC[e];
      float rg = sigmoidf_(gir + sGh[r * G3_ + c]);
      float zg = sigmoidf_(giz + sGh[r * G3_ + 64 + c]);
      float ng = tanhf(gin + rg * (sGh[r * G3_ + 128 + c] + sBhn[c]));
      float nv = (1.f - zg) * ng + zg * cv;
      sC[e] = nv;
      hs[(((size_t)t * B_ + bb) * N_ + nn) * H_ + c] = (bf16)nv;  // bf16 hs
    }
    __syncthreads();
  }
#pragma unroll
  for (int i = 0; i < 4; ++i) {
    int e = tid + 256 * i;
    int r = e >> 6, c = e & 63;
    hfin[(size_t)(row0 + r) * H_ + c] = sC[e];
  }
}

// ------- kernel 2: per-factor MHA (16 factors = 48 rows / block) -------
__global__ __launch_bounds__(256)
void attn_kernel(const bf16* __restrict__ hs, const int* __restrict__ indices,
                 const float* __restrict__ Wq, const float* __restrict__ bq,
                 const float* __restrict__ Wk, const float* __restrict__ bk,
                 const float* __restrict__ Wv, const float* __restrict__ bv,
                 const float* __restrict__ Wo, const float* __restrict__ bo,
                 float* __restrict__ y) {
  constexpr int FB = 16;                       // factors per block
  constexpr int RR = FB * ORD_;                // 48 rows = 3 A panels
  __shared__ alignas(32) bf16  sW[3][H_ * H_]; // swizzled Wq,Wk,Wv
  __shared__ alignas(32) bf16  sWo[H_ * H_];   // swizzled
  __shared__ float sB[3][H_];
  __shared__ float sBo[H_];
  __shared__ alignas(32) bf16  sG[RR * H_];    // swizzled A panels; reused for O
  __shared__ bf16  sQKV[3][RR * H_];           // linear (attention VALU reads)
  __shared__ int   sNode[RR];

  const int tid = threadIdx.x, lane = tid & 31, wave = tid >> 5;
  const int nfb = F_ / FB;                     // 1024
  const int fb = blockIdx.x % nfb;
  const int tb = blockIdx.x / nfb;             // t*B + b
  const int f0 = fb * FB;

  const float* Ws[3] = {Wq, Wk, Wv};
  const float* Bs[3] = {bq, bk, bv};
#pragma unroll
  for (int m = 0; m < 3; ++m) {
    for (int i = tid; i < H_ * H_; i += 256) {
      int k = i >> 6, n = i & 63;
      sW[m][b_swz(k, n, H_)] = (bf16)Ws[m][i];
    }
    for (int i = tid; i < H_; i += 256) sB[m][i] = Bs[m][i];
  }
  for (int i = tid; i < H_ * H_; i += 256) {
    int k = i >> 6, n = i & 63;
    sWo[b_swz(k, n, H_)] = (bf16)Wo[i];
  }
  for (int i = tid; i < H_; i += 256) sBo[i] = bo[i];
  if (tid < RR) sNode[tid] = indices[(f0 + tid / ORD_) * ORD_ + tid % ORD_];
  __syncthreads();

  // -------- async gather: hs rows -> swizzled bf16 A panels in LDS --------
  // Each bf16 row is 128B; every aligned 16B chunk of a row maps to one
  // contiguous 16B LDS range in the A-fragment swizzle, so the gather is
  // per-lane GLOBAL_LOAD_ASYNC_TO_LDS_B128 (ASYNCcnt-tracked, no VGPR data).
  {
    const unsigned sg_base = (unsigned)(size_t)(lds_bf16*)sG;
    for (int j = tid; j < RR * 8; j += 256) {  // 48 rows x 8 chunks
      int fr = j >> 3, ch = j & 7;
      int m = fr & 15, panel = fr >> 4;
      unsigned dst = sg_base +
          2u * (unsigned)(panel * 1024 + (ch >> 2) * 512 +
                          (m + ((ch & 1) << 4)) * 16 + (((ch >> 1) & 1) << 3));
      const bf16* src = hs + ((size_t)tb * N_ + sNode[fr]) * H_ + ch * 8;
      asm volatile("global_load_async_to_lds_b128 %0, %1, off"
                   :: "v"(dst), "v"(src) : "memory");
    }
    asm volatile("s_wait_asynccnt 0x0" ::: "memory");
  }
  __syncthreads();

  // Q,K,V projections: 3 mats x 3 row-tiles x 4 col-tiles = 36 WMMA jobs
  for (int j = wave; j < 36; j += 8) {
    int m = j / 12, rem = j % 12, rt = rem / 4, ct = rem % 4;
    float8 acc = f8zero();
#pragma unroll
    for (int ks = 0; ks < 64; ks += 32) {
      bhalf16 a = ldsA_swz(sG + rt * 1024, ks >> 5, lane);
      bhalf16 b = ldsB_swz(sW[m], ks >> 5, ct, H_, lane);
      acc = wmma_bf16(a, b, acc);
    }
    int n = lane & 15, mb = (lane < 16) ? 0 : 8;
    float bias = sB[m][ct * 16 + n];
#pragma unroll
    for (int i = 0; i < 8; ++i)
      sQKV[m][(rt * 16 + mb + i) * H_ + ct * 16 + n] = (bf16)(acc[i] + bias);
  }
  __syncthreads();
  // 3x3 softmax-attention per (factor, head); O overwrites sG (swizzled)
  if (tid < FB * NH_) {
    int fl = tid >> 2, hh = tid & 3;
    int fr0 = fl * 3, off = hh * HD_;
    float p[3][3];
#pragma unroll
    for (int i = 0; i < 3; ++i) {
      float mx = -1e30f;
#pragma unroll
      for (int jq = 0; jq < 3; ++jq) {
        float acc = 0.f;
#pragma unroll
        for (int d = 0; d < HD_; ++d)
          acc += (float)sQKV[0][(fr0 + i) * H_ + off + d] *
                 (float)sQKV[1][(fr0 + jq) * H_ + off + d];
        p[i][jq] = acc * 0.25f;                // 1/sqrt(16)
        mx = fmaxf(mx, p[i][jq]);
      }
      float sum = 0.f;
#pragma unroll
      for (int jq = 0; jq < 3; ++jq) { p[i][jq] = __expf(p[i][jq] - mx); sum += p[i][jq]; }
      float inv = 1.f / sum;
#pragma unroll
      for (int jq = 0; jq < 3; ++jq) p[i][jq] *= inv;
    }
#pragma unroll
    for (int i = 0; i < 3; ++i)
#pragma unroll
      for (int d = 0; d < HD_; ++d) {
        float o = 0.f;
#pragma unroll
        for (int jq = 0; jq < 3; ++jq)
          o += p[i][jq] * (float)sQKV[2][(fr0 + jq) * H_ + off + d];
        int fr = fr0 + i;
        sG[(fr >> 4) * 1024 + a_swz(fr & 15, off + d)] = (bf16)o;
      }
  }
  __syncthreads();
  // output projection + atomic scatter-add into y
  for (int j = wave; j < 12; j += 8) {
    int rt = j / 4, ct = j % 4;
    float8 acc = f8zero();
#pragma unroll
    for (int ks = 0; ks < 64; ks += 32) {
      bhalf16 a = ldsA_swz(sG + rt * 1024, ks >> 5, lane);
      bhalf16 b = ldsB_swz(sWo, ks >> 5, ct, H_, lane);
      acc = wmma_bf16(a, b, acc);
    }
    int n = lane & 15, mb = (lane < 16) ? 0 : 8;
    float bias = sBo[ct * 16 + n];
#pragma unroll
    for (int i = 0; i < 8; ++i) {
      int fr = rt * 16 + mb + i;
      int node = sNode[fr];
      atomicAdd(&y[((size_t)tb * N_ + node) * H_ + ct * 16 + n], acc[i] + bias);
    }
  }
}

// ----- kernel 3: y /= nnz, block-reduce partition means into yg -----
__global__ __launch_bounds__(256)
void norm_reduce_kernel(float* __restrict__ y, const float* __restrict__ nnz,
                        float* __restrict__ yg) {
  __shared__ float sRed[256];
  size_t e = (size_t)blockIdx.x * 256 + threadIdx.x;
  int c = (int)(e & 63);
  int n = (int)((e >> 6) % N_);
  int tb = (int)(e / ((size_t)N_ * H_));
  float v = y[e] / nnz[n];
  y[e] = v;
  sRed[threadIdx.x] = v;
  __syncthreads();
  if (threadIdx.x < 64) {                      // 4 consecutive n share (t,b,k,c)
    float s = sRed[threadIdx.x] + sRed[threadIdx.x + 64] +
              sRed[threadIdx.x + 128] + sRed[threadIdx.x + 192];
    int k = n >> 12;                           // n / N0_, uniform within block
    atomicAdd(&yg[((size_t)tb * KALL_ + k) * H_ + c], s * (1.0f / N0_));
  }
}

// ---------------- kernel 4: value head (tiny) ----------------
__global__ void value_kernel(const float* __restrict__ yg,
                             const float* __restrict__ extra,
                             const float* __restrict__ Wv1,
                             const float* __restrict__ bv1,
                             float* __restrict__ val) {
  __shared__ float sv[T_ * B_];
  int tid = threadIdx.x;
  if (tid < T_ * B_) sv[tid] = 0.f;
  __syncthreads();
  if (tid < T_ * B_ * KALL_) {
    int tb = tid >> 2, k = tid & 3;
    float s = bv1[0];
    for (int j = 0; j < H_; ++j)
      s += yg[((size_t)tb * KALL_ + k) * H_ + j] * Wv1[j];
    for (int j = 0; j < DE_; ++j)
      s += extra[((size_t)tb * KALL_ + k) * DE_ + j] * Wv1[H_ + j];
    atomicAdd(&sv[tb], s);
  }
  __syncthreads();
  if (tid < T_ * B_) val[tid] = sv[tid];
}

// --------- kernel 5: logit head (64 rows / block, K padded 68->96) ---------
__global__ __launch_bounds__(256)
void logit_kernel(const float* __restrict__ y, const float* __restrict__ extra,
                  const float* __restrict__ W1, const float* __restrict__ b1,
                  const float* __restrict__ W2, const float* __restrict__ b2,
                  float* __restrict__ logit) {
  constexpr int RPB = 64, KP = 96;
  __shared__ alignas(32) bf16  sW1[KP * DOUT_];  // swizzled (rows 68..95 zero)
  __shared__ float sB1[DOUT_], sW2[DOUT_];
  __shared__ alignas(32) bf16  sZ[RPB * KP];     // 4 swizzled A panels
  __shared__ float sR[RPB * DOUT_];
  __shared__ float sB2;

  const int tid = threadIdx.x, lane = tid & 31, wave = tid >> 5;
  const int r0 = blockIdx.x * RPB;

  for (int i = tid; i < KP * DOUT_; i += 256) {
    int kk = i / DOUT_, c = i % DOUT_;
    sW1[b_swz(kk, c, DOUT_)] = (bf16)((kk < H_ + DE_) ? W1[kk * DOUT_ + c] : 0.f);
  }
  for (int i = tid; i < DOUT_; i += 256) { sB1[i] = b1[i]; sW2[i] = W2[i]; }
  if (tid == 0) sB2 = b2[0];
  for (int e = tid; e < RPB * KP; e += 256) {
    int lr = e / KP, c = e % KP;
    int gr = r0 + lr;
    int n = gr % N_, tb = gr / N_;
    float v = 0.f;
    if (c < H_)            v = y[(size_t)gr * H_ + c];
    else if (c < H_ + DE_) v = extra[((size_t)tb * KALL_ + (n >> 12)) * DE_ + (c - H_)];
    sZ[(lr >> 4) * (16 * KP) + a_swz(lr & 15, c)] = (bf16)v;
  }
  __syncthreads();
  {
    int rt = wave >> 1;                          // 4 row-tiles, 2 waves each
    int ct0 = (wave & 1) * 2;
    for (int ct = ct0; ct < ct0 + 2; ++ct) {
      float8 acc = f8zero();
#pragma unroll
      for (int ks = 0; ks < KP; ks += 32) {
        bhalf16 a = ldsA_swz(sZ + rt * (16 * KP), ks >> 5, lane);
        bhalf16 b = ldsB_swz(sW1, ks >> 5, ct, DOUT_, lane);
        acc = wmma_bf16(a, b, acc);
      }
      int n = lane & 15, mb = (lane < 16) ? 0 : 8;
#pragma unroll
      for (int i = 0; i < 8; ++i) {
        float v = acc[i] + sB1[ct * 16 + n];
        sR[(rt * 16 + mb + i) * DOUT_ + ct * 16 + n] = fmaxf(v, 0.f);
      }
    }
  }
  __syncthreads();
  if (tid < RPB) {
    float s = sB2;
    for (int c = 0; c < DOUT_; ++c) s += sR[tid * DOUT_ + c] * sW2[c];
    logit[(size_t)(r0 + tid)] = s;
  }
}

// ---------------- kernel 6: passthrough copy ----------------
__global__ void copy_kernel(const float* __restrict__ src, float* __restrict__ dst, int n) {
  int i = blockIdx.x * blockDim.x + threadIdx.x;
  if (i < n) dst[i] = src[i];
}

// ---------------- host launch ----------------
extern "C" void kernel_launch(void* const* d_in, const int* in_sizes, int n_in,
                              void* d_out, int out_size, void* d_ws, size_t ws_size,
                              hipStream_t stream) {
  (void)in_sizes; (void)n_in; (void)out_size; (void)ws_size;
  const float* h0     = (const float*)d_in[0];
  const float* hglob  = (const float*)d_in[1];
  const float* x      = (const float*)d_in[2];
  const float* extra  = (const float*)d_in[3];
  const unsigned char* dones = (const unsigned char*)d_in[4];
  const int*   indices = (const int*)d_in[5];
  const float* nnz    = (const float*)d_in[6];
  const float* Wi  = (const float*)d_in[7];
  const float* bi  = (const float*)d_in[8];
  const float* Wh  = (const float*)d_in[9];
  const float* bhn = (const float*)d_in[10];
  const float* Wq  = (const float*)d_in[11];
  const float* bq  = (const float*)d_in[12];
  const float* Wk  = (const float*)d_in[13];
  const float* bk  = (const float*)d_in[14];
  const float* Wv  = (const float*)d_in[15];
  const float* bv  = (const float*)d_in[16];
  const float* Wo  = (const float*)d_in[17];
  const float* bo  = (const float*)d_in[18];
  const float* W1  = (const float*)d_in[19];
  const float* b1  = (const float*)d_in[20];
  const float* W2  = (const float*)d_in[21];
  const float* b2  = (const float*)d_in[22];
  const float* Wv1 = (const float*)d_in[23];
  const float* bv1 = (const float*)d_in[24];

  // output layout: h_fin | h_global | logit | value
  float* out       = (float*)d_out;
  float* out_hfin  = out;
  float* out_hg    = out_hfin + (size_t)B_ * N_ * H_;
  float* out_logit = out_hg + (size_t)B_ * KALL_ * DOUT_;
  float* out_val   = out_logit + (size_t)T_ * B_ * N_;

  // workspace: hs[T,B,N,H] (bf16) | y[T,B,N,H] (f32) | yg[T,B,K,H] (f32)
  bf16*  hsb = (bf16*)d_ws;
  float* y   = (float*)(hsb + (size_t)T_ * B_ * N_ * H_);
  float* yg  = y + (size_t)T_ * B_ * N_ * H_;

  const size_t nzero = (size_t)T_ * B_ * N_ * H_ + (size_t)T_ * B_ * KALL_ * H_;
  zero_kernel<<<4096, 256, 0, stream>>>(y, nzero);   // y and yg are contiguous

  gru_kernel<<<(B_ * N_) / 16, 256, 0, stream>>>(h0, x, dones, Wi, bi, Wh, bhn,
                                                 hsb, out_hfin);

  attn_kernel<<<T_ * B_ * (F_ / 16), 256, 0, stream>>>(hsb, indices,
                                                       Wq, bq, Wk, bk, Wv, bv,
                                                       Wo, bo, y);

  norm_reduce_kernel<<<(int)(((size_t)T_ * B_ * N_ * H_) / 256), 256, 0, stream>>>(
      y, nnz, yg);

  value_kernel<<<1, 64, 0, stream>>>(yg, extra, Wv1, bv1, out_val);

  logit_kernel<<<(T_ * B_ * N_) / 64, 256, 0, stream>>>(y, extra, W1, b1, W2, b2,
                                                        out_logit);

  copy_kernel<<<(B_ * KALL_ * DOUT_ + 255) / 256, 256, 0, stream>>>(
      hglob, out_hg, B_ * KALL_ * DOUT_);
}
